// SEA_15857019257347
// MI455X (gfx1250) — compile-verified, matrix operand order checked by reading, not run on hardware
//
#include <hip/hip_runtime.h>

#define BN_TOK (256 * 512)
#define TDIM 96
#define HDIM 128
#define EXPERTS 8

typedef __attribute__((ext_vector_type(16))) __bf16 v16bf;
typedef __attribute__((ext_vector_type(8)))  float  v8f;
typedef __attribute__((ext_vector_type(8)))  unsigned short us8;
typedef __attribute__((ext_vector_type(4)))  float  f32x4v;
typedef __attribute__((ext_vector_type(4)))  __bf16 bf16x4;

typedef __attribute__((ext_vector_type(4))) unsigned int u32x4;
typedef __attribute__((ext_vector_type(8))) int i32x8;
typedef __attribute__((ext_vector_type(4))) int i32x4;

union Frag16 {
    v16bf v;
    us8   u[2];
    unsigned short s[16];
};

// Native f32 -> bf16 (RTNE) conversions; lowers to v_cvt(_pk)_bf16_f32 on gfx1250.
__device__ __forceinline__ unsigned short f2bf(float f) {
    union { __bf16 h; unsigned short s; } u; u.h = (__bf16)f; return u.s;
}
__device__ __forceinline__ void cvt4(unsigned short* dst, float4 c) {
    f32x4v v; v[0] = c.x; v[1] = c.y; v[2] = c.z; v[3] = c.w;
    union { bf16x4 b; unsigned short s[4]; } u;
    u.b = __builtin_convertvector(v, bf16x4);
    dst[0] = u.s[0]; dst[1] = u.s[1]; dst[2] = u.s[2]; dst[3] = u.s[3];
}

// ---------------------------------------------------------------------------
// TDM: 1-row 2D descriptor copying `units8 * 8` contiguous bytes from global
// memory into LDS at byte offset `lds_off`. Issued by one wave; completion
// via s_wait_tensorcnt. (CDNA5 ISA)
// ---------------------------------------------------------------------------
#if defined(__gfx1250__)
__device__ __forceinline__ void tdm_load_blob(unsigned lds_off, const void* gptr,
                                              unsigned units8) {
    unsigned long long ga = (unsigned long long)(size_t)gptr;
    u32x4 g0;
    g0[0] = 1u;                                               // count=1, user D#
    g0[1] = lds_off;                                          // lds_addr (bytes)
    g0[2] = (unsigned)ga;                                     // global_addr[31:0]
    g0[3] = (unsigned)((ga >> 32) & 0x1FFFFFFu) | (2u << 30); // addr[56:32] | type=2
    i32x8 g1;
    g1[0] = (int)(3u << 16);                                  // wg_mask=0, data_size=3 (8B)
    g1[1] = (int)((units8 & 0xFFFFu) << 16);                  // tensor_dim0[15:0]
    g1[2] = (int)(((units8 >> 16) & 0xFFFFu) | (1u << 16));   // dim0 hi | tensor_dim1=1
    g1[3] = (int)((units8 & 0xFFFFu) << 16);                  // dim1 hi=0 | tile_dim0
    g1[4] = 1;                                                // tile_dim1=1, tile_dim2=0
    g1[5] = (int)units8;                                      // tensor_dim0_stride lo
    g1[6] = 0;                                                // stride0 hi | stride1 lo
    g1[7] = 0;                                                // stride1 hi
    i32x4 z4 = {0, 0, 0, 0};
#if __clang_major__ >= 23
    i32x8 z8 = {0, 0, 0, 0, 0, 0, 0, 0};
    __builtin_amdgcn_tensor_load_to_lds(g0, g1, z4, z4, z8, 0);
#else
    __builtin_amdgcn_tensor_load_to_lds(g0, g1, z4, z4, 0);
#endif
}
#endif

// ---------------------------------------------------------------------------
// Kernel 1: zero routing counters + repack W1/W2 (fp32) into bf16 fragments
// laid out in WMMA B-matrix per-lane order:  [e][ks][nt][lane][16 bf16]
// ---------------------------------------------------------------------------
__global__ void SEA_prep(const float* __restrict__ W1, const float* __restrict__ W2,
                         unsigned short* __restrict__ w1f, unsigned short* __restrict__ w2f,
                         int* __restrict__ cnt)
{
    if (blockIdx.x == 0 && threadIdx.x < EXPERTS) cnt[threadIdx.x] = 0;
    const int NW1 = EXPERTS * 3 * 8 * 32 * 16;   // 98304
    const int NW2 = EXPERTS * 4 * 6 * 32 * 16;   // 98304
    int f = blockIdx.x * blockDim.x + threadIdx.x;
    if (f < NW1) {
        int elem = f & 15, lane = (f >> 4) & 31, rest = f >> 9;
        int nt = rest & 7; rest >>= 3;
        int ks = rest % 3, e = rest / 3;
        int g = lane >> 4, n = lane & 15;
        int K = 32 * ks + ((elem < 8) ? (elem + 8 * g) : (elem + 8 + 8 * g));
        w1f[f] = f2bf(W1[((e * TDIM) + K) * HDIM + nt * 16 + n]);
    } else if (f < NW1 + NW2) {
        int f2 = f - NW1;
        int elem = f2 & 15, lane = (f2 >> 4) & 31, rest = f2 >> 9;
        int nt = rest % 6; rest /= 6;
        int ks = rest & 3, e = rest >> 2;
        int g = lane >> 4, n = lane & 15;
        int K = 32 * ks + ((elem < 8) ? (elem + 8 * g) : (elem + 8 + 8 * g));
        w2f[f2] = f2bf(W2[((e * HDIM) + K) * TDIM + nt * 16 + n]);
    }
}

// ---------------------------------------------------------------------------
// Kernel 2: gating. One thread per token: logits -> softmax -> top-2 ->
// threshold mask -> normalized weights. Writes util, out = route_scale*x,
// and appends (token, route_scale*wn) to per-expert routed lists.
// ---------------------------------------------------------------------------
__global__ void SEA_gate(const float* __restrict__ x, const float* __restrict__ Wg,
                         const float* __restrict__ bg, const float* __restrict__ rsp,
                         float* __restrict__ out, float* __restrict__ util,
                         int* __restrict__ cnt, int* __restrict__ toklist,
                         float* __restrict__ coeflist)
{
    int i = blockIdx.x * blockDim.x + threadIdx.x;
    if (i >= BN_TOK) return;
    const float rs = rsp[0];
    const float* xr = x + (size_t)i * TDIM;

    float lg[EXPERTS];
#pragma unroll
    for (int e = 0; e < EXPERTS; ++e) lg[e] = bg[e];
    for (int t = 0; t < TDIM; t += 4) {
        float4 xv = *(const float4*)(xr + t);
#pragma unroll
        for (int e = 0; e < EXPERTS; ++e) {
            const float* wr = Wg + e * TDIM + t;
            lg[e] += xv.x * wr[0] + xv.y * wr[1] + xv.z * wr[2] + xv.w * wr[3];
        }
    }
    float mx = -1e30f;
#pragma unroll
    for (int e = 0; e < EXPERTS; ++e) { lg[e] *= 0.1f; mx = fmaxf(mx, lg[e]); }
    float p[EXPERTS]; float sum = 0.f;
#pragma unroll
    for (int e = 0; e < EXPERTS; ++e) { p[e] = __expf(lg[e] - mx); sum += p[e]; }
    float inv = 1.f / sum;
#pragma unroll
    for (int e = 0; e < EXPERTS; ++e) p[e] *= inv;

    int i0 = 0; float p0 = p[0];
#pragma unroll
    for (int e = 1; e < EXPERTS; ++e) if (p[e] > p0) { p0 = p[e]; i0 = e; }
    int i1 = -1; float p1 = -1.f;
#pragma unroll
    for (int e = 0; e < EXPERTS; ++e) if (e != i0 && p[e] > p1) { p1 = p[e]; i1 = e; }

    float s  = p0 + p1;
    float w0 = p0 / (s + 1e-12f), w1 = p1 / (s + 1e-12f);
    bool  m0 = (w0 >= 0.05f), m1 = (w1 >= 0.05f);
    float invsw = 1.f / fmaxf(w0 + w1, 1e-12f);
    float wn0 = w0 * invsw, wn1 = w1 * invsw;

    // out = route_scale * x  (residual + unmasked fallback, since sum(wn)==1)
    for (int t = 0; t < TDIM; t += 4) {
        float4 xv = *(const float4*)(xr + t);
        float4 ov; ov.x = rs * xv.x; ov.y = rs * xv.y; ov.z = rs * xv.z; ov.w = rs * xv.w;
        *(float4*)(out + (size_t)i * TDIM + t) = ov;
    }
#pragma unroll
    for (int e = 0; e < EXPERTS; ++e) {
        float u = ((m0 && e == i0) ? 1.f : 0.f) + ((m1 && e == i1) ? 1.f : 0.f);
        util[(size_t)i * EXPERTS + e] = u;
    }
    if (m0) { int q = atomicAdd(&cnt[i0], 1); toklist[(size_t)i0 * BN_TOK + q] = i; coeflist[(size_t)i0 * BN_TOK + q] = rs * wn0; }
    if (m1) { int q = atomicAdd(&cnt[i1], 1); toklist[(size_t)i1 * BN_TOK + q] = i; coeflist[(size_t)i1 * BN_TOK + q] = rs * wn1; }
}

// ---------------------------------------------------------------------------
// Kernel 3: routed expert MLP with bf16 WMMA + TDM weight staging.
// Block = 8 waves; each wave owns 16 routed tokens. B fragments are
// register-double-buffered so ds_load latency overlaps the WMMA pipe.
// ---------------------------------------------------------------------------
__global__ __launch_bounds__(256) void SEA_expert(
    const float* __restrict__ x,
    const unsigned short* __restrict__ w1f,
    const unsigned short* __restrict__ w2f,
    const float* __restrict__ b1,
    const float* __restrict__ b2,
    const int* __restrict__ cnt,
    const int* __restrict__ toklist,
    const float* __restrict__ coeflist,
    float* __restrict__ out)
{
    const int e = blockIdx.y;
    const int count = cnt[e];
    const int tileBase = blockIdx.x * 128;
    if (tileBase >= count) return;                 // uniform per block

    __shared__ __align__(16) unsigned short hbuf[8][16][HDIM]; // 32 KB
    __shared__ __align__(16) unsigned short sW[12288];         // 24 KB (W1 then W2)
    __shared__ int   stok[8][16];
    __shared__ float scoef[8][16];

    const int lane = threadIdx.x & 31;
    const int wave = threadIdx.x >> 5;
    const int g = lane >> 4;
    const int n = lane & 15;

    const int rowBase = tileBase + wave * 16;
    if (lane < 16) {
        int idx = rowBase + lane;
        int tok = 0; float cf = 0.f;
        if (idx < count) { tok = toklist[(size_t)e * BN_TOK + idx]; cf = coeflist[(size_t)e * BN_TOK + idx]; }
        stok[wave][lane]  = tok;
        scoef[wave][lane] = cf;
    }

    // --- Stage W1[e] fragments into LDS (async tensor DMA, wave 0 only) ---
#if defined(__gfx1250__)
    if (wave == 0) {
        tdm_load_blob((unsigned)(size_t)(void*)sW, w1f + (size_t)e * 12288, 12288 * 2 / 8);
        __builtin_amdgcn_s_wait_tensorcnt(0);
    }
#else
    for (unsigned idx = threadIdx.x; idx < 12288; idx += 256)
        sW[idx] = w1f[(size_t)e * 12288 + idx];
#endif
    __syncthreads();

    // A fragments (16-bit A layout): lane group g holds K = k0+8g+{0..7} and
    // K = k0+16+8g+{0..7} of row m = lane%16. Convert fp32 x -> bf16 on the fly.
    const int myTok = stok[wave][n];
    const float* xrow = x + (size_t)myTok * TDIM;
    Frag16 a1[3];
#pragma unroll
    for (int ks = 0; ks < 3; ++ks) {
        cvt4(&a1[ks].s[0],  *(const float4*)(xrow + 32 * ks + 8 * g));
        cvt4(&a1[ks].s[4],  *(const float4*)(xrow + 32 * ks + 8 * g + 4));
        cvt4(&a1[ks].s[8],  *(const float4*)(xrow + 32 * ks + 16 + 8 * g));
        cvt4(&a1[ks].s[12], *(const float4*)(xrow + 32 * ks + 16 + 8 * g + 4));
    }

    // Prefetch W2 fragments toward L2 while GEMM1 runs (global_prefetch_b8).
    __builtin_prefetch(w2f + ((size_t)e * 4 * 6 * 32 + lane) * 16, 0, 1);

    // ---- GEMM1: h = relu(x @ W1[e] + b1[e]), B-frags double-buffered ----
    Frag16 bc[4], bn[4];
#pragma unroll
    for (int ks = 0; ks < 3; ++ks) {
        const unsigned short* bp = sW + (((ks * 8 + 0) * 32 + lane) << 4);
        bc[ks].u[0] = *(const us8*)(bp);
        bc[ks].u[1] = *(const us8*)(bp + 8);
    }
#pragma unroll
    for (int nt = 0; nt < 8; ++nt) {
        if (nt < 7) {
#pragma unroll
            for (int ks = 0; ks < 3; ++ks) {
                const unsigned short* bp = sW + (((ks * 8 + nt + 1) * 32 + lane) << 4);
                bn[ks].u[0] = *(const us8*)(bp);
                bn[ks].u[1] = *(const us8*)(bp + 8);
            }
        }
        v8f acc = {};
#pragma unroll
        for (int ks = 0; ks < 3; ++ks)
            acc = __builtin_amdgcn_wmma_f32_16x16x32_bf16(false, a1[ks].v, false, bc[ks].v,
                                                          (short)0, acc, false, false);
        const float bias = b1[e * HDIM + nt * 16 + n];
#pragma unroll
        for (int v = 0; v < 8; ++v) {
            float hv = fmaxf(acc[v] + bias, 0.f);     // C/D layout: M = v+8g, N = n
            hbuf[wave][v + 8 * g][nt * 16 + n] = f2bf(hv);
        }
#pragma unroll
        for (int ks = 0; ks < 3; ++ks) bc[ks] = bn[ks];
    }
    __syncthreads();   // h visible; all W1 reads complete -> sW reusable

    // --- Stage W2[e] fragments into the same LDS buffer ---
#if defined(__gfx1250__)
    if (wave == 0) {
        tdm_load_blob((unsigned)(size_t)(void*)sW, w2f + (size_t)e * 12288, 12288 * 2 / 8);
        __builtin_amdgcn_s_wait_tensorcnt(0);
    }
#else
    for (unsigned idx = threadIdx.x; idx < 12288; idx += 256)
        sW[idx] = w2f[(size_t)e * 12288 + idx];
#endif

    // ---- GEMM2 A fragments from staged h (per-wave private region) ----
    Frag16 a2[4];
#pragma unroll
    for (int ks = 0; ks < 4; ++ks) {
        a2[ks].u[0] = *(const us8*)(&hbuf[wave][n][32 * ks + 8 * g]);
        a2[ks].u[1] = *(const us8*)(&hbuf[wave][n][32 * ks + 16 + 8 * g]);
    }
    __syncthreads();   // W2 staged and visible to all waves

    // ---- GEMM2: y = h @ W2[e] + b2[e];  out += coef * y ----
#pragma unroll
    for (int ks = 0; ks < 4; ++ks) {
        const unsigned short* bp = sW + (((ks * 6 + 0) * 32 + lane) << 4);
        bc[ks].u[0] = *(const us8*)(bp);
        bc[ks].u[1] = *(const us8*)(bp + 8);
    }
#pragma unroll
    for (int nt = 0; nt < 6; ++nt) {
        if (nt < 5) {
#pragma unroll
            for (int ks = 0; ks < 4; ++ks) {
                const unsigned short* bp = sW + (((ks * 6 + nt + 1) * 32 + lane) << 4);
                bn[ks].u[0] = *(const us8*)(bp);
                bn[ks].u[1] = *(const us8*)(bp + 8);
            }
        }
        v8f acc = {};
#pragma unroll
        for (int ks = 0; ks < 4; ++ks)
            acc = __builtin_amdgcn_wmma_f32_16x16x32_bf16(false, a2[ks].v, false, bc[ks].v,
                                                          (short)0, acc, false, false);
        const float bias = b2[e * TDIM + nt * 16 + n];
#pragma unroll
        for (int v = 0; v < 8; ++v) {
            const int m = v + 8 * g;
            float val = (acc[v] + bias) * scoef[wave][m];
            atomicAdd(out + (size_t)stok[wave][m] * TDIM + nt * 16 + n, val);
        }
#pragma unroll
        for (int ks = 0; ks < 4; ++ks) bc[ks] = bn[ks];
    }
}

// ---------------------------------------------------------------------------
extern "C" void kernel_launch(void* const* d_in, const int* in_sizes, int n_in,
                              void* d_out, int out_size, void* d_ws, size_t ws_size,
                              hipStream_t stream)
{
    const float* x  = (const float*)d_in[0];
    const float* Wg = (const float*)d_in[1];
    const float* bg = (const float*)d_in[2];
    const float* W1 = (const float*)d_in[3];
    const float* b1 = (const float*)d_in[4];
    const float* W2 = (const float*)d_in[5];
    const float* b2 = (const float*)d_in[6];
    const float* rs = (const float*)d_in[7];

    float* out  = (float*)d_out;
    float* util = out + (size_t)BN_TOK * TDIM;

    char* ws = (char*)d_ws;
    int*   cnt      = (int*)ws;                                    // 32 B (pad 256)
    int*   toklist  = (int*)(ws + 256);                            // 4 MB
    float* coeflist = (float*)(ws + 256 + 4u * 1024 * 1024);       // 4 MB
    unsigned short* w1f = (unsigned short*)(ws + 256 + 8u * 1024 * 1024); // 192 KB
    unsigned short* w2f = w1f + EXPERTS * 3 * 8 * 32 * 16;                // 192 KB

    SEA_prep<<<768, 256, 0, stream>>>(W1, W2, w1f, w2f, cnt);
    SEA_gate<<<BN_TOK / 256, 256, 0, stream>>>(x, Wg, bg, rs, out, util,
                                               cnt, toklist, coeflist);
    dim3 egrid(BN_TOK / 128, EXPERTS);
    SEA_expert<<<egrid, 256, 0, stream>>>(x, w1f, w2f, b1, b2,
                                          cnt, toklist, coeflist, out);
    (void)in_sizes; (void)n_in; (void)out_size; (void)ws_size;
}
